// Sparse_EGT_Layer_7009386627596
// MI455X (gfx1250) — matrix operand order repaired and checked by reading, not compiled, not run on hardware
//
#include <hip/hip_runtime.h>
#include <math.h>

#define NN 100000
#define EE 1600000

typedef __attribute__((ext_vector_type(16))) __bf16 v16bf;
typedef __attribute__((ext_vector_type(8)))  float  v8f;

__device__ __forceinline__ unsigned short f2bf(float f) {
  unsigned int u = __float_as_uint(f);
  unsigned int r = u + 0x7FFFu + ((u >> 16) & 1u);   // round-to-nearest-even
  return (unsigned short)(r >> 16);
}
__device__ __forceinline__ __bf16 bfbits(unsigned short s) {
  return __builtin_bit_cast(__bf16, s);
}

__device__ __forceinline__ v8f wmma_bf16(v16bf a, v16bf b, v8f c) {
  return __builtin_amdgcn_wmma_f32_16x16x32_bf16(false, a, false, b, (short)0, c, false, false);
}

// A fragment (16x32 bf16) from row-major LDS tile [16][ld].
// Lane L<16: row L, K in {hi*8+0..7} (vgpr pairs 0..3) and {16+hi*8+0..7} (pairs 4..7)
__device__ __forceinline__ v16bf load_a(const unsigned short* base, int ld, int kBase, int lane) {
  int m  = lane & 15;
  int hi = lane >> 4;
  const unsigned short* row = base + m * ld + kBase + hi * 8;
  v16bf a;
#pragma unroll
  for (int j = 0; j < 16; ++j) {
    int k = ((j < 8) ? 0 : 16) + (j & 7);
    a[j] = bfbits(row[k]);
  }
  return a;
}

// B fragment (32x16 bf16), B[K][N] = W[N][K] with W row-major [out][in] stride ld.
// slot j -> K = kBase + 8*hi + (j<8 ? j : 16 + (j-8)) ; N = nBase + (lane&15)
__device__ __forceinline__ v16bf load_b(const unsigned short* w, int ld, int nBase, int kBase, int lane) {
  int n  = nBase + (lane & 15);
  int k0 = kBase + ((lane >> 4) << 3);
  const unsigned short* row = w + n * ld + k0;
  v16bf b;
#pragma unroll
  for (int j = 0; j < 16; ++j) b[j] = bfbits(row[((j < 8) ? 0 : 16) + (j & 7)]);
  return b;
}

__device__ __forceinline__ float gelu_exact(float x) {
  return 0.5f * x * (1.0f + erff(x * 0.70710678118654752f));
}

// LayerNorm over 64 cols held as 4 C-tiles x[t][r]; row = r+8*hi lives on a 16-lane group.
__device__ __forceinline__ void ln_rows(float x[4][8], const float* g, const float* b, int nlo) {
#pragma unroll
  for (int r = 0; r < 8; ++r) {
    float s = x[0][r] + x[1][r] + x[2][r] + x[3][r];
#pragma unroll
    for (int off = 1; off < 16; off <<= 1) s += __shfl_xor(s, off, 32);
    float mu = s * (1.0f / 64.0f);
    float var = 0.f;
#pragma unroll
    for (int t = 0; t < 4; ++t) { float d = x[t][r] - mu; var += d * d; }
#pragma unroll
    for (int off = 1; off < 16; off <<= 1) var += __shfl_xor(var, off, 32);
    float inv = rsqrtf(var * (1.0f / 64.0f) + 1e-5f);
#pragma unroll
    for (int t = 0; t < 4; ++t) {
      int col = t * 16 + nlo;
      x[t][r] = (x[t][r] - mu) * inv * g[col] + b[col];
    }
  }
}

__device__ __forceinline__ void atomAddF(float* p, float v) {
  __hip_atomic_fetch_add(p, v, __ATOMIC_RELAXED, __HIP_MEMORY_SCOPE_AGENT);
}

// ---------------------------------------------------------------- kernel 1: QKV
__global__ __launch_bounds__(128) void qkv_kernel(
    const float* __restrict__ h_node,
    const float* __restrict__ Wq, const float* __restrict__ bq,
    const float* __restrict__ Wk, const float* __restrict__ bk,
    const float* __restrict__ Wv, const float* __restrict__ bv,
    float* __restrict__ qo, float* __restrict__ ko, float* __restrict__ vo) {
  __shared__ unsigned short WqB[64 * 64], WkB[64 * 64], WvB[64 * 64];
  __shared__ float biases[3 * 64];
  __shared__ unsigned short st[4][16 * 64];
  int tid = threadIdx.x;
  for (int i = tid; i < 64 * 64; i += 128) {
    WqB[i] = f2bf(Wq[i]); WkB[i] = f2bf(Wk[i]); WvB[i] = f2bf(Wv[i]);
  }
  for (int i = tid; i < 64; i += 128) {
    biases[i] = bq[i]; biases[64 + i] = bk[i]; biases[128 + i] = bv[i];
  }
  __syncthreads();

  int w = tid >> 5, lane = tid & 31;
  int tt = blockIdx.x * 4 + w;
  if (tt >= NN / 16) return;           // wave-uniform
  int n0 = tt * 16;
  int nlo = lane & 15, hi = lane >> 4;

  for (int idx = lane; idx < 16 * 16; idx += 32) {
    int row = idx >> 4, c4 = idx & 15;
    float4 f = *(const float4*)(h_node + (size_t)(n0 + row) * 64 + c4 * 4);
    unsigned short* d = &st[w][row * 64 + c4 * 4];
    d[0] = f2bf(f.x); d[1] = f2bf(f.y); d[2] = f2bf(f.z); d[3] = f2bf(f.w);
  }
  v16bf a0 = load_a(st[w], 64, 0, lane);
  v16bf a1 = load_a(st[w], 64, 32, lane);

  const unsigned short* Ws[3] = {WqB, WkB, WvB};
  float* outs[3] = {qo, ko, vo};
#pragma unroll
  for (int mi = 0; mi < 3; ++mi) {
#pragma unroll
    for (int nt = 0; nt < 4; ++nt) {
      v8f c = {};
      c = wmma_bf16(a0, load_b(Ws[mi], 64, nt * 16, 0, lane), c);
      c = wmma_bf16(a1, load_b(Ws[mi], 64, nt * 16, 32, lane), c);
      int col = nt * 16 + nlo;
      float bias = biases[mi * 64 + col];
#pragma unroll
      for (int r = 0; r < 8; ++r) {
        int row = n0 + r + 8 * hi;
        outs[mi][(size_t)row * 64 + col] = c[r] + bias;
      }
    }
  }
}

// ------------------------------------------------- kernel 2: score + seg-max
__global__ __launch_bounds__(256) void score_kernel(
    const float* __restrict__ q, const float* __restrict__ k,
    const float* __restrict__ h_edge,
    const int* __restrict__ esrc, const int* __restrict__ edst,
    const float* __restrict__ Web, const float* __restrict__ beb,
    float* __restrict__ score, unsigned int* __restrict__ segmax) {
  __shared__ float WebS[8 * 64];
  __shared__ float bebS[8];
  int tid = threadIdx.x;
  for (int i = tid; i < 8 * 64; i += 256) WebS[i] = Web[i];
  if (tid < 8) bebS[tid] = beb[tid];
  __syncthreads();
  size_t e = (size_t)blockIdx.x * 256 + tid;
  if (e >= EE) return;
  int s = esrc[e], d = edst[e];

  float bias[8];
#pragma unroll
  for (int h = 0; h < 8; ++h) bias[h] = bebS[h];
  const float* he = h_edge + e * 64;
  for (int c = 0; c < 64; ++c) {
    float x = he[c];
#pragma unroll
    for (int h = 0; h < 8; ++h) bias[h] += WebS[h * 64 + c] * x;
  }
  const float* qr = q + (size_t)d * 64;
  const float* kr = k + (size_t)s * 64;
#pragma unroll
  for (int h = 0; h < 8; ++h) {
    float acc = 0.f;
#pragma unroll
    for (int j = 0; j < 8; ++j) acc += qr[h * 8 + j] * kr[h * 8 + j];
    float sc = acc * 0.35355339059327373f + bias[h];
    score[e * 8 + h] = sc;
    unsigned int u = __float_as_uint(sc);
    u = (u & 0x80000000u) ? ~u : (u | 0x80000000u);  // order-preserving map
    atomicMax(&segmax[(size_t)d * 8 + h], u);
  }
}

// ------------------------------------------------- kernel 3: exp + seg-sum
__global__ __launch_bounds__(256) void ex_kernel(
    float* __restrict__ score, const unsigned int* __restrict__ segmax,
    float* __restrict__ segsum, const int* __restrict__ edst) {
  size_t e = (size_t)blockIdx.x * 256 + threadIdx.x;
  if (e >= EE) return;
  int d = edst[e];
#pragma unroll
  for (int h = 0; h < 8; ++h) {
    unsigned int u = segmax[(size_t)d * 8 + h];
    float m = (u & 0x80000000u) ? __uint_as_float(u & 0x7FFFFFFFu)
                                : __uint_as_float(~u);
    float exv = expf(score[e * 8 + h] - m);
    score[e * 8 + h] = exv;
    atomAddF(&segsum[(size_t)d * 8 + h], exv);
  }
}

// ------------------------------------------------- kernel 4: scatter messages
__global__ __launch_bounds__(256) void scatter_kernel(
    const float* __restrict__ v, const float* __restrict__ ex,
    const float* __restrict__ segsum,
    const int* __restrict__ esrc, const int* __restrict__ edst,
    float* __restrict__ agg) {
  size_t gid = (size_t)blockIdx.x * 256 + threadIdx.x;
  if (gid >= (size_t)EE * 64) return;
  size_t e = gid >> 6;
  int c = (int)(gid & 63);
  int h = c >> 3;
  int s = esrc[e], d = edst[e];
  float prob = ex[e * 8 + h] / segsum[(size_t)d * 8 + h];
  atomAddF(&agg[(size_t)d * 64 + c], v[(size_t)s * 64 + c] * prob);
}

// ------------------------- kernel 5: Wo GEMM + LN1 + FFN + LN2 (fused, WMMA)
__global__ __launch_bounds__(128) void node_post_kernel(
    const float* __restrict__ h_node, const float* __restrict__ agg,
    const float* __restrict__ Wo, const float* __restrict__ bo,
    const float* __restrict__ ln1g, const float* __restrict__ ln1b,
    const float* __restrict__ Wf1, const float* __restrict__ bf1,
    const float* __restrict__ Wf2, const float* __restrict__ bf2,
    const float* __restrict__ ln2g, const float* __restrict__ ln2b,
    float* __restrict__ hout) {
  __shared__ unsigned short WoB[64 * 64], Wf1B[128 * 64], Wf2B[64 * 128];
  __shared__ float pb[512];  // bo|ln1g|ln1b|bf2|ln2g|ln2b|bf1(128)
  __shared__ unsigned short st[4][16 * 128];
  int tid = threadIdx.x;
  for (int i = tid; i < 64 * 64; i += 128) WoB[i] = f2bf(Wo[i]);
  for (int i = tid; i < 128 * 64; i += 128) { Wf1B[i] = f2bf(Wf1[i]); Wf2B[i] = f2bf(Wf2[i]); }
  for (int i = tid; i < 64; i += 128) {
    pb[i] = bo[i]; pb[64 + i] = ln1g[i]; pb[128 + i] = ln1b[i];
    pb[192 + i] = bf2[i]; pb[256 + i] = ln2g[i]; pb[320 + i] = ln2b[i];
  }
  pb[384 + tid] = bf1[tid];
  __syncthreads();

  int w = tid >> 5, lane = tid & 31;
  int tt = blockIdx.x * 4 + w;
  if (tt >= NN / 16) return;
  int n0 = tt * 16;
  int nlo = lane & 15, hi = lane >> 4;

  // stage agg tile as bf16
  for (int idx = lane; idx < 16 * 16; idx += 32) {
    int row = idx >> 4, c4 = idx & 15;
    float4 f = *(const float4*)(agg + (size_t)(n0 + row) * 64 + c4 * 4);
    unsigned short* dp = &st[w][row * 128 + c4 * 4];
    dp[0] = f2bf(f.x); dp[1] = f2bf(f.y); dp[2] = f2bf(f.z); dp[3] = f2bf(f.w);
  }
  v16bf a0 = load_a(st[w], 128, 0, lane);
  v16bf a1 = load_a(st[w], 128, 32, lane);

  float x[4][8];
#pragma unroll
  for (int nt = 0; nt < 4; ++nt) {
    v8f c = {};
    c = wmma_bf16(a0, load_b(WoB, 64, nt * 16, 0, lane), c);
    c = wmma_bf16(a1, load_b(WoB, 64, nt * 16, 32, lane), c);
    int col = nt * 16 + nlo;
#pragma unroll
    for (int r = 0; r < 8; ++r) {
      int row = n0 + r + 8 * hi;
      x[nt][r] = c[r] + pb[col] + h_node[(size_t)row * 64 + col];  // +bo +residual
    }
  }
  ln_rows(x, &pb[64], &pb[128], nlo);  // x = h1

  // store h1 bf16 for FFN1 A-fragments
#pragma unroll
  for (int nt = 0; nt < 4; ++nt)
#pragma unroll
    for (int r = 0; r < 8; ++r)
      st[w][(r + 8 * hi) * 128 + nt * 16 + nlo] = f2bf(x[nt][r]);
  a0 = load_a(st[w], 128, 0, lane);
  a1 = load_a(st[w], 128, 32, lane);

  float f1[8][8];
#pragma unroll
  for (int nt = 0; nt < 8; ++nt) {
    v8f c = {};
    c = wmma_bf16(a0, load_b(Wf1B, 64, nt * 16, 0, lane), c);
    c = wmma_bf16(a1, load_b(Wf1B, 64, nt * 16, 32, lane), c);
    int col = nt * 16 + nlo;
#pragma unroll
    for (int r = 0; r < 8; ++r) f1[nt][r] = gelu_exact(c[r] + pb[384 + col]);
  }
  // store gelu(FFN1) as 16x128 bf16
#pragma unroll
  for (int nt = 0; nt < 8; ++nt)
#pragma unroll
    for (int r = 0; r < 8; ++r)
      st[w][(r + 8 * hi) * 128 + nt * 16 + nlo] = f2bf(f1[nt][r]);
  v16bf aa0 = load_a(st[w], 128, 0, lane);
  v16bf aa1 = load_a(st[w], 128, 32, lane);
  v16bf aa2 = load_a(st[w], 128, 64, lane);
  v16bf aa3 = load_a(st[w], 128, 96, lane);

#pragma unroll
  for (int nt = 0; nt < 4; ++nt) {
    v8f c = {};
    c = wmma_bf16(aa0, load_b(Wf2B, 128, nt * 16, 0, lane), c);
    c = wmma_bf16(aa1, load_b(Wf2B, 128, nt * 16, 32, lane), c);
    c = wmma_bf16(aa2, load_b(Wf2B, 128, nt * 16, 64, lane), c);
    c = wmma_bf16(aa3, load_b(Wf2B, 128, nt * 16, 96, lane), c);
    int col = nt * 16 + nlo;
#pragma unroll
    for (int r = 0; r < 8; ++r)
      x[nt][r] = x[nt][r] + c[r] + pb[192 + col];  // h1 + ffn + bf2
  }
  ln_rows(x, &pb[256], &pb[320], nlo);
#pragma unroll
  for (int nt = 0; nt < 4; ++nt) {
    int col = nt * 16 + nlo;
#pragma unroll
    for (int r = 0; r < 8; ++r)
      hout[(size_t)(n0 + r + 8 * hi) * 64 + col] = x[nt][r];
  }
}

// ---------------------------- kernel 6: edge MLP (gather + 192-K GEMM, WMMA)
__global__ __launch_bounds__(128) void edge_mlp_kernel(
    const float* __restrict__ hnew, const float* __restrict__ h_edge,
    const int* __restrict__ esrc, const int* __restrict__ edst,
    const float* __restrict__ We1, const float* __restrict__ be1,
    const float* __restrict__ We2, const float* __restrict__ be2,
    const float* __restrict__ lneg, const float* __restrict__ lneb,
    float* __restrict__ he_out) {
  __shared__ unsigned short We1B[64 * 192], We2B[64 * 64];
  __shared__ float pb[256];  // be1|be2|lneg|lneb
  __shared__ unsigned short st[4][16 * 192];
  int tid = threadIdx.x;
  for (int i = tid; i < 64 * 192; i += 128) We1B[i] = f2bf(We1[i]);
  for (int i = tid; i < 64 * 64; i += 128) We2B[i] = f2bf(We2[i]);
  for (int i = tid; i < 64; i += 128) {
    pb[i] = be1[i]; pb[64 + i] = be2[i]; pb[128 + i] = lneg[i]; pb[192 + i] = lneb[i];
  }
  __syncthreads();

  int w = tid >> 5, lane = tid & 31;
  int te = blockIdx.x * 4 + w;  // EE/16 == 100000, grid*4 exact
  size_t e0 = (size_t)te * 16;
  int nlo = lane & 15, hi = lane >> 4;

  // gather X = [h[src] | h[dst] | h_edge] -> bf16 LDS [16][192]
  for (int idx = lane; idx < 16 * 48; idx += 32) {
    int row = idx / 48, c4 = idx % 48;
    size_t e = e0 + row;
    const float* src;
    if (c4 < 16)       src = hnew + (size_t)esrc[e] * 64 + c4 * 4;
    else if (c4 < 32)  src = hnew + (size_t)edst[e] * 64 + (c4 - 16) * 4;
    else               src = h_edge + e * 64 + (c4 - 32) * 4;
    float4 f = *(const float4*)src;
    unsigned short* dp = &st[w][row * 192 + c4 * 4];
    dp[0] = f2bf(f.x); dp[1] = f2bf(f.y); dp[2] = f2bf(f.z); dp[3] = f2bf(f.w);
  }
  v16bf a[6];
#pragma unroll
  for (int kc = 0; kc < 6; ++kc) a[kc] = load_a(st[w], 192, kc * 32, lane);

  float g[4][8];
#pragma unroll
  for (int nt = 0; nt < 4; ++nt) {
    v8f c = {};
#pragma unroll
    for (int kc = 0; kc < 6; ++kc)
      c = wmma_bf16(a[kc], load_b(We1B, 192, nt * 16, kc * 32, lane), c);
    int col = nt * 16 + nlo;
#pragma unroll
    for (int r = 0; r < 8; ++r) g[nt][r] = gelu_exact(c[r] + pb[col]);
  }
  // stage gelu output (16x64) back into st for GEMM2
#pragma unroll
  for (int nt = 0; nt < 4; ++nt)
#pragma unroll
    for (int r = 0; r < 8; ++r)
      st[w][(r + 8 * hi) * 192 + nt * 16 + nlo] = f2bf(g[nt][r]);
  v16bf ga0 = load_a(st[w], 192, 0, lane);
  v16bf ga1 = load_a(st[w], 192, 32, lane);

  float z[4][8];
#pragma unroll
  for (int nt = 0; nt < 4; ++nt) {
    v8f c = {};
    c = wmma_bf16(ga0, load_b(We2B, 64, nt * 16, 0, lane), c);
    c = wmma_bf16(ga1, load_b(We2B, 64, nt * 16, 32, lane), c);
    int col = nt * 16 + nlo;
#pragma unroll
    for (int r = 0; r < 8; ++r) {
      size_t row = e0 + r + 8 * hi;
      z[nt][r] = c[r] + pb[64 + col] + h_edge[row * 64 + col];  // +be2 +residual
    }
  }
  ln_rows(z, &pb[128], &pb[192], nlo);
#pragma unroll
  for (int nt = 0; nt < 4; ++nt) {
    int col = nt * 16 + nlo;
#pragma unroll
    for (int r = 0; r < 8; ++r) {
      size_t row = e0 + r + 8 * hi;
      he_out[row * 64 + col] = z[nt][r];
    }
  }
}

// ---------------------------------------------------------------------------
extern "C" void kernel_launch(void* const* d_in, const int* in_sizes, int n_in,
                              void* d_out, int out_size, void* d_ws, size_t ws_size,
                              hipStream_t stream) {
  const float* h_node = (const float*)d_in[0];
  const float* h_edge = (const float*)d_in[1];
  const int*   eidx   = (const int*)d_in[2];
  const int*   esrc   = eidx;        // edge_index[0]
  const int*   edst   = eidx + EE;   // edge_index[1]
  const float* Wq  = (const float*)d_in[3];   const float* bq  = (const float*)d_in[4];
  const float* Wk  = (const float*)d_in[5];   const float* bk  = (const float*)d_in[6];
  const float* Wv  = (const float*)d_in[7];   const float* bv  = (const float*)d_in[8];
  const float* Wo  = (const float*)d_in[9];   const float* bo  = (const float*)d_in[10];
  const float* Web = (const float*)d_in[11];  const float* beb = (const float*)d_in[12];
  const float* ln1g = (const float*)d_in[13]; const float* ln1b = (const float*)d_in[14];
  const float* ln2g = (const float*)d_in[15]; const float* ln2b = (const float*)d_in[16];
  const float* lneg = (const float*)d_in[17]; const float* lneb = (const float*)d_in[18];
  const float* Wf1 = (const float*)d_in[19];  const float* bf1 = (const float*)d_in[20];
  const float* Wf2 = (const float*)d_in[21];  const float* bf2 = (const float*)d_in[22];
  const float* We1 = (const float*)d_in[23];  const float* be1 = (const float*)d_in[24];
  const float* We2 = (const float*)d_in[25];  const float* be2 = (const float*)d_in[26];

  float* q   = (float*)d_ws;
  float* k   = q + (size_t)NN * 64;
  float* v   = k + (size_t)NN * 64;
  float* sc  = v + (size_t)NN * 64;                       // E*8, reused as ex
  unsigned int* segmax = (unsigned int*)(sc + (size_t)EE * 8);
  float* segsum = (float*)(segmax + (size_t)NN * 8);
  float* agg    = segsum + (size_t)NN * 8;

  hipMemsetAsync(segmax, 0, (size_t)NN * 8 * 4, stream);
  hipMemsetAsync(segsum, 0, (size_t)NN * 8 * 4, stream);
  hipMemsetAsync(agg,    0, (size_t)NN * 64 * 4, stream);

  float* hout  = (float*)d_out;
  float* heout = hout + (size_t)NN * 64;

  int node_blocks = (NN / 16 + 3) / 4;       // 1563
  qkv_kernel<<<node_blocks, 128, 0, stream>>>(h_node, Wq, bq, Wk, bk, Wv, bv, q, k, v);
  score_kernel<<<EE / 256, 256, 0, stream>>>(q, k, h_edge, esrc, edst, Web, beb, sc, segmax);
  ex_kernel<<<EE / 256, 256, 0, stream>>>(sc, segmax, segsum, edst);
  scatter_kernel<<<(unsigned)((size_t)EE * 64 / 256), 256, 0, stream>>>(v, sc, segsum, esrc, edst, agg);
  node_post_kernel<<<node_blocks, 128, 0, stream>>>(h_node, agg, Wo, bo, ln1g, ln1b,
                                                    Wf1, bf1, Wf2, bf2, ln2g, ln2b, hout);
  edge_mlp_kernel<<<EE / 16 / 4, 128, 0, stream>>>(hout, h_edge, esrc, edst,
                                                   We1, be1, We2, be2, lneg, lneb, heout);
}
